// Model_85160611545156
// MI455X (gfx1250) — matrix operand order repaired and checked by reading, not compiled, not run on hardware
//
#include <hip/hip_runtime.h>
#include <stdint.h>
#include <stddef.h>

// Model dims (match reference)
#define B_    256
#define S_    336
#define P_    96
#define H_    512
#define G4H   2048          // 4*H
#define TOUT  (S_ + P_ - 1) // 431

typedef __attribute__((ext_vector_type(16))) __bf16        v16bf;
typedef __attribute__((ext_vector_type(8)))  float         v8f;
typedef __attribute__((ext_vector_type(8)))  unsigned int  v8u;

__device__ __forceinline__ unsigned short f2bf(float f) {
  unsigned u = __float_as_uint(f);
  u += 0x7fffu + ((u >> 16) & 1u);   // round-to-nearest-even
  return (unsigned short)(u >> 16);
}
// Branch-free transcendentals: native v_exp_f32 + v_rcp_f32, no exec-mask
// divergence (keeps the wave EXEC=all-ones schedule clean around WMMA).
__device__ __forceinline__ float rcp_(float x) { return __builtin_amdgcn_rcpf(x); }
__device__ __forceinline__ float sigmoidf_(float x) { return rcp_(1.f + __expf(-x)); }
__device__ __forceinline__ float tanhf_(float x) {
  float t = __expf(-2.f * fabsf(x));
  float r = (1.f - t) * rcp_(1.f + t);
  return copysignf(r, x);
}

// ---------------------------------------------------------------------------
// Pack Wih/Whh (both branches) f32 [2048,512] -> bf16 WMMA-B tiles.
// B tile (K=32,N=16) layout: n_local = lane&15, halfword s = k_local in table
// order (s<8 ? 8*(lane>>4)+s : 16+8*(lane>>4)+s-8); one 32B chunk per lane per tile.
// halfword index = nt*8192 + kt*512 + lane*16 + s ; B[k][n] = W[n][k].
// ---------------------------------------------------------------------------
__global__ __launch_bounds__(256) void k_pack(const float* __restrict__ w0,
                                              const float* __restrict__ w1,
                                              const float* __restrict__ w2,
                                              const float* __restrict__ w3,
                                              unsigned short* __restrict__ pw) {
  unsigned gid  = blockIdx.x * 256u + threadIdx.x;   // [0, 4*1048576)
  unsigned mat  = gid >> 20;
  unsigned rem  = gid & 0xFFFFFu;
  unsigned s    = rem & 15u;
  unsigned lane = (rem >> 4) & 31u;
  unsigned kt   = (rem >> 9) & 15u;
  unsigned nt   = rem >> 13;                         // 0..127
  unsigned n = nt * 16u + (lane & 15u);              // gate column (row of W)
  unsigned hb = lane >> 4;
  unsigned k = kt * 32u + (s < 8u ? 8u * hb + s : 16u + 8u * hb + (s - 8u)); // hidden index (col of W)
  const float* w = (mat == 0) ? w0 : (mat == 1) ? w1 : (mat == 2) ? w2 : w3;
  pw[gid] = f2bf(w[(size_t)n * H_ + k]);
}

// ---------------------------------------------------------------------------
// Fold encoder into the recurrence:  Wc = Wih @ We  [2048 x cin],
// bEnc = bih + bhh + Wih @ be,  bPred = bih + bhh.
// ---------------------------------------------------------------------------
__global__ __launch_bounds__(256) void k_misc(
    const float* __restrict__ linWe,  const float* __restrict__ linBe,
    const float* __restrict__ quatWe, const float* __restrict__ quatBe,
    const float* __restrict__ linWih, const float* __restrict__ linBih,
    const float* __restrict__ linBhh,
    const float* __restrict__ quatWih,const float* __restrict__ quatBih,
    const float* __restrict__ quatBhh,
    float* __restrict__ wc, float* __restrict__ bEnc, float* __restrict__ bPred) {
  int gid = blockIdx.x * 256 + threadIdx.x;          // [0, 2*2048)
  int br = gid >> 11, r = gid & 2047;
  const float* We  = br ? quatWe  : linWe;
  const float* be  = br ? quatBe  : linBe;
  const float* Wih = br ? quatWih : linWih;
  const float* bih = br ? quatBih : linBih;
  const float* bhh = br ? quatBhh : linBhh;
  int cin = br ? 4 : 3;
  float wcv[4] = {0.f, 0.f, 0.f, 0.f};
  float bx = 0.f;
  for (int h = 0; h < H_; ++h) {
    float w = Wih[(size_t)r * H_ + h];
    bx += w * be[h];
    for (int j = 0; j < cin; ++j) wcv[j] += w * We[h * cin + j];
  }
  for (int j = 0; j < 4; ++j) wc[gid * 4 + j] = wcv[j];
  float bp = bih[r] + bhh[r];
  bPred[gid] = bp;
  bEnc[gid]  = bp + bx;
}

// ---------------------------------------------------------------------------
// Persistent LSTM kernel: 32 blocks = 2 branches x 16 batch-groups of 16 rows.
// Per step: G[16,2048] = pre + h@W^T via v_wmma_f32_16x16x32_bf16, then fused
// gates + c/h update + on-the-fly decode to d_out.
// ---------------------------------------------------------------------------
struct __align__(32) SM {
  float          G[16][G4H];          // 128 KB  gate pre-activations
  unsigned short hA[16 * 32 * 16];    //  16 KB  h in WMMA-A layout (kt,lane,s)
  float          wd[4 * H_];          //   8 KB  decoder weights
  float          xs[16 * 4];          //  256 B  x tile (padded to 4)
};
static_assert(sizeof(SM) <= 160 * 1024, "LDS budget");

__global__ __launch_bounds__(256) void k_lstm(
    const float* __restrict__ x, const unsigned short* __restrict__ packAll,
    const float* __restrict__ wcAll, const float* __restrict__ bEncAll,
    const float* __restrict__ bPredAll,
    const float* __restrict__ linDw, const float* __restrict__ linDb,
    const float* __restrict__ quatDw, const float* __restrict__ quatDb,
    float* __restrict__ out) {
  __shared__ SM sm;
  const int tid  = threadIdx.x;
  const int lane = tid & 31, wave = tid >> 5;
  const int br   = blockIdx.x >> 4;      // 0=linear, 1=quaternion branch
  const int grp  = blockIdx.x & 15;      // batch group (16 rows)
  const int cin  = br ? 4 : 3;
  const int cOff = br ? 3 : 0;
  const int dout = br ? 4 : 3;

  const unsigned short* pwIh = packAll + (size_t)(br * 2 + 0) * (G4H * H_);
  const unsigned short* pwHh = packAll + (size_t)(br * 2 + 1) * (G4H * H_);
  const float* wc = wcAll   + (size_t)br * G4H * 4;
  const float* bE = bEncAll + (size_t)br * G4H;
  const float* bP = bPredAll+ (size_t)br * G4H;
  const float* dW = br ? quatDw : linDw;
  const float* dB = br ? quatDb : linDb;

  for (int i = tid; i < 16 * 32 * 16; i += 256) sm.hA[i] = 0;     // h0 = 0
  for (int i = tid; i < dout * H_; i += 256) sm.wd[i] = dW[i];
  float creg[32];
  #pragma unroll
  for (int q = 0; q < 32; ++q) creg[q] = 0.f;                     // c0 = 0

  const int hlf = lane >> 4;   // C-layout half: M = v + 8*hlf
  const int col = lane & 15;   // C-layout column within N-tile

  for (int st = 0; st < S_ + P_; ++st) {
    const bool enc = st < S_;
    if (enc && tid < 64) {
      int m = tid >> 2, j = tid & 3;
      sm.xs[tid] = (j < cin)
          ? x[((size_t)(grp * 16 + m) * S_ + st) * 7 + cOff + j] : 0.f;
    }
    __syncthreads();   // hA from previous step + xs ready

    // --- matmul phase: wave covers N-tiles [wave*16, wave*16+16) ---
    // Cache the full A strip (h tile, 16 k-tiles) in registers once per step:
    // 128 VGPRs, removes 16x redundant LDS reads across the 16 N-tiles.
    v16bf Areg[16];
    #pragma unroll
    for (int kt = 0; kt < 16; ++kt)
      Areg[kt] = __builtin_bit_cast(v16bf, *(const v8u*)&sm.hA[(kt * 32 + lane) * 16]);

    const unsigned short* pw = enc ? pwHh : pwIh;   // pred feeds h into Wih
    for (int i = 0; i < 16; ++i) {
      int nt = wave * 16 + i;
      int r  = nt * 16 + col;                       // gate column of this lane
      v8f acc;
      float bias = enc ? bE[r] : bP[r];
      if (enc) {                                    // pre = x·Wc + bias
        float4 wv = *(const float4*)(wc + r * 4);
        #pragma unroll
        for (int v = 0; v < 8; ++v) {
          float4 xv = *(const float4*)&sm.xs[(v + hlf * 8) * 4];
          acc[v] = bias + xv.x * wv.x + xv.y * wv.y + xv.z * wv.z + xv.w * wv.w;
        }
      } else {
        #pragma unroll
        for (int v = 0; v < 8; ++v) acc[v] = bias;
      }
      const unsigned short* pb = pw + ((size_t)(nt * 16) * 32 + lane) * 16;
      #pragma unroll
      for (int kt = 0; kt < 16; ++kt) {
        v16bf Bv = __builtin_bit_cast(v16bf, *(const v8u*)(pb + kt * (32 * 16)));
        acc = __builtin_amdgcn_wmma_f32_16x16x32_bf16(
            false, Areg[kt], false, Bv, (short)0, acc, false, false);
      }
      #pragma unroll
      for (int v = 0; v < 8; ++v) sm.G[v + hlf * 8][r] = acc[v];
    }
    __syncthreads();   // G complete

    // --- fused gates + state update + decode partials ---
    const int row = tid >> 4;
    const int jb  = (tid & 15) << 5;   // 32-wide column block of H
    float pd0 = 0.f, pd1 = 0.f, pd2 = 0.f, pd3 = 0.f;
    #pragma unroll
    for (int q = 0; q < 32; ++q) {
      int j = jb + q;
      float gi = sm.G[row][j];
      float gf = sm.G[row][j + 512];
      float gg = sm.G[row][j + 1024];
      float go = sm.G[row][j + 1536];
      float cp = enc ? creg[q] : 0.f;            // pred steps: zero init state
      float cn = sigmoidf_(gf) * cp + sigmoidf_(gi) * tanhf_(gg);
      float hn = sigmoidf_(go) * tanhf_(cn);
      creg[q] = cn;
      pd0 += hn * sm.wd[j];
      pd1 += hn * sm.wd[j + 512];
      pd2 += hn * sm.wd[j + 1024];
      if (dout == 4) pd3 += hn * sm.wd[j + 1536];
      // write h back in WMMA-A layout: k_local = (s<8 ? s : s+8) + 8*half
      int kt = j >> 5, kl = j & 31;
      int hf = (kl >> 3) & 1;
      int s  = kl - hf * 8 - ((kl >= 16) ? 8 : 0);
      sm.hA[(kt * 32 + (hf * 16 + row)) * 16 + s] = f2bf(hn);
    }
    // reduce 16 threads of each row (xor<16 stays within 16-lane half)
    #pragma unroll
    for (int m = 1; m < 16; m <<= 1) {
      pd0 += __shfl_xor(pd0, m, 32);
      pd1 += __shfl_xor(pd1, m, 32);
      pd2 += __shfl_xor(pd2, m, 32);
      pd3 += __shfl_xor(pd3, m, 32);
    }
    int out_t = st - 1;                           // seq drops first step
    if (((tid & 15) == 0) && out_t >= 0) {
      float* op = out + ((size_t)(grp * 16 + row) * TOUT + out_t) * 7 + cOff;
      op[0] = pd0 + dB[0];
      op[1] = pd1 + dB[1];
      op[2] = pd2 + dB[2];
      if (dout == 4) op[3] = pd3 + dB[3];
    }
    __syncthreads();   // hA writes done before next matmul reads
  }
}

// ---------------------------------------------------------------------------
extern "C" void kernel_launch(void* const* d_in, const int* in_sizes, int n_in,
                              void* d_out, int out_size, void* d_ws, size_t ws_size,
                              hipStream_t stream) {
  (void)in_sizes; (void)n_in; (void)out_size; (void)ws_size;
  const float* x        = (const float*)d_in[0];
  const float* lin_We   = (const float*)d_in[1];
  const float* lin_be   = (const float*)d_in[2];
  const float* quat_We  = (const float*)d_in[3];
  const float* quat_be  = (const float*)d_in[4];
  const float* lin_Wih  = (const float*)d_in[5];
  const float* lin_Whh  = (const float*)d_in[6];
  const float* lin_bih  = (const float*)d_in[7];
  const float* lin_bhh  = (const float*)d_in[8];
  const float* quat_Wih = (const float*)d_in[9];
  const float* quat_Whh = (const float*)d_in[10];
  const float* quat_bih = (const float*)d_in[11];
  const float* quat_bhh = (const float*)d_in[12];
  const float* lin_Dw   = (const float*)d_in[13];
  const float* lin_Db   = (const float*)d_in[14];
  const float* quat_Dw  = (const float*)d_in[15];
  const float* quat_Db  = (const float*)d_in[16];
  // d_in[17] = pred_len (96, compile-time constant P_)

  // workspace layout
  unsigned short* pw = (unsigned short*)d_ws;                // 8 MB packed bf16
  const size_t PACK_BYTES = 4ull * G4H * H_ * sizeof(unsigned short);
  float* wc    = (float*)((char*)d_ws + PACK_BYTES);         // 2*2048*4 f32
  float* bEnc  = wc + 2 * G4H * 4;
  float* bPred = bEnc + 2 * G4H;

  k_pack<<<dim3(4u * 1024u * 1024u / 256u), dim3(256), 0, stream>>>(
      lin_Wih, lin_Whh, quat_Wih, quat_Whh, pw);
  k_misc<<<dim3(2 * G4H / 256), dim3(256), 0, stream>>>(
      lin_We, lin_be, quat_We, quat_be,
      lin_Wih, lin_bih, lin_bhh,
      quat_Wih, quat_bih, quat_bhh,
      wc, bEnc, bPred);
  k_lstm<<<dim3(32), dim3(256), 0, stream>>>(
      x, pw, wc, bEnc, bPred,
      lin_Dw, lin_Db, quat_Dw, quat_Db, (float*)d_out);
}
